// GraphAttentionLayer_22376779612748
// MI455X (gfx1250) — matrix operand order repaired and checked by reading, not compiled, not run on hardware
//
#include <hip/hip_runtime.h>
#include <stdint.h>

typedef _Float16 half_t;
typedef __attribute__((ext_vector_type(16))) _Float16 v16h;
typedef __attribute__((ext_vector_type(8)))  float    v8f;
typedef __attribute__((ext_vector_type(4)))  unsigned int u32x4;
typedef __attribute__((ext_vector_type(8)))  int      i32x8;
typedef __attribute__((ext_vector_type(4)))  int      i32x4;
typedef __attribute__((ext_vector_type(4)))  float    f32x4;

#define B_    8
#define N_    2048
#define INF_  256
#define OUTF_ 256
#define ALPHA_ 0.2f
#define NEGINF_ (-1e9f)

union AB32 { u32x4 u[2]; v16h h; };   // 32 bytes = one WMMA 16-bit operand per lane
union H8  { _Float16 h[8]; u32x4 u; };

#if __has_builtin(__builtin_amdgcn_tensor_load_to_lds)
#define HAVE_TDM 1
#else
#define HAVE_TDM 0
#endif

// Issue a TDM load of WhT[b][f=0..255][m0..m0+31] (16 KB, 2-byte elems) into LDS.
// 2D tensor: dim0 = n (len 2048, stride 2048), dim1 = f (len 256); tile 32 x 256.
__device__ __forceinline__ void stage_issue(half_t* dst, const half_t* src, int lane) {
#if HAVE_TDM
    unsigned lds_byte     = (unsigned)(uintptr_t)dst;        // LDS aperture: addr[31:0]
    unsigned long long ga = (unsigned long long)(uintptr_t)src;
    u32x4 g0;
    g0[0] = 1u;                                              // count=1 (valid user D#)
    g0[1] = lds_byte;                                        // lds_addr
    g0[2] = (unsigned)ga;                                    // global_addr[31:0]
    g0[3] = (unsigned)((ga >> 32) & 0x01FFFFFFu) | (2u << 30); // addr[56:32] | type=2
    i32x8 g1;
    g1[0] = 0x00010000;                                      // data_size=1 (2 bytes)
    g1[1] = (int)(((unsigned)N_) << 16);                     // tensor_dim0[15:0] @bits[63:48]
    g1[2] = (int)(((unsigned)OUTF_) << 16);                  // dim0 hi=0, tensor_dim1 lo
    g1[3] = (int)(32u << 16);                                // dim1 hi=0, tile_dim0=32
    g1[4] = OUTF_;                                           // tile_dim1=256, tile_dim2=0
    g1[5] = N_;                                              // tensor_dim0_stride = 2048
    g1[6] = 0;
    g1[7] = 0;
    i32x4 z = {0, 0, 0, 0};
#if __clang_major__ >= 23
    i32x8 z8 = {0, 0, 0, 0, 0, 0, 0, 0};
    __builtin_amdgcn_tensor_load_to_lds(g0, g1, z, z, z8, 0);
#else
    __builtin_amdgcn_tensor_load_to_lds(g0, g1, z, z, 0);
#endif
#else
    // fallback: wave0's 32 lanes copy 256 rows x 64B cooperatively
#pragma unroll
    for (int f0 = 0; f0 < 8; ++f0) {
        int f = f0 * 32 + lane;
        const u32x4* s = (const u32x4*)(src + (size_t)f * N_);
        u32x4* d = (u32x4*)(dst + f * 32);
        d[0] = s[0]; d[1] = s[1]; d[2] = s[2]; d[3] = s[3];
    }
#endif
}

__device__ __forceinline__ void stage_wait() {
#if HAVE_TDM
    __builtin_amdgcn_s_wait_tensorcnt(0);
#else
    __threadfence_block();
#endif
}

// ---------------- kernel 0: fp32 -> f16 conversion of h and W ----------------
__global__ void cvt_kernel(const float* __restrict__ h, const float* __restrict__ W,
                           half_t* __restrict__ h16, half_t* __restrict__ W16) {
    int i = blockIdx.x * blockDim.x + threadIdx.x;
    if (i < B_ * N_ * INF_) h16[i] = (half_t)h[i];
    if (i < OUTF_ * INF_)   W16[i] = (half_t)W[i];
}

// ---------------- kernel 1: Wh = h @ W^T + b, stored transposed f16 ----------
__global__ void __launch_bounds__(256) wh_kernel(const half_t* __restrict__ h16,
                                                 const half_t* __restrict__ W16,
                                                 const float*  __restrict__ Wb,
                                                 half_t* __restrict__ whT) {
    const int lane = threadIdx.x & 31;
    const int wave = threadIdx.x >> 5;
    const int g    = lane >> 4;
    const int ln   = lane & 15;

    const int tile = blockIdx.x * 8 + wave;
    const int rt = tile >> 4;
    const int ct = tile & 15;
    const int b  = rt >> 7;
    const int n0 = (rt & 127) << 4;
    const int o0 = ct << 4;

    const half_t* aptr = h16 + (size_t)(b * N_ + n0 + ln) * INF_;
    const half_t* bptr = W16 + (size_t)(o0 + ln) * INF_;

    v8f c = {};
#pragma unroll
    for (int k0 = 0; k0 < INF_; k0 += 32) {
        AB32 A, Bm;
        A.u[0]  = *(const u32x4*)(aptr + k0 + g * 8);
        A.u[1]  = *(const u32x4*)(aptr + k0 + 16 + g * 8);
        Bm.u[0] = *(const u32x4*)(bptr + k0 + g * 8);
        Bm.u[1] = *(const u32x4*)(bptr + k0 + 16 + g * 8);
        c = __builtin_amdgcn_wmma_f32_16x16x32_f16(false, A.h, false, Bm.h,
                                                   (short)0, c, false, false);
    }
    const float bias = Wb[o0 + ln];
    H8 st;
#pragma unroll
    for (int r = 0; r < 8; ++r) st.h[r] = (half_t)(c[r] + bias);
    *(u32x4*)(whT + (size_t)(b * OUTF_ + o0 + ln) * N_ + n0 + g * 8) = st.u;
}

// ---------------- kernel 2: attention score halves s1, s2 --------------------
__global__ void score_kernel(const half_t* __restrict__ whT,
                             const float* __restrict__ ai_w, const float* __restrict__ ai_b,
                             const float* __restrict__ aj_w, const float* __restrict__ aj_b,
                             float* __restrict__ s1, float* __restrict__ s2) {
    int i = blockIdx.x * blockDim.x + threadIdx.x;
    int b = i >> 11, n = i & (N_ - 1);
    const half_t* base = whT + (size_t)b * OUTF_ * N_ + n;
    float a1 = 0.f, a2 = 0.f;
#pragma unroll 4
    for (int f = 0; f < OUTF_; ++f) {
        float w = (float)base[(size_t)f * N_];
        a1 += w * ai_w[f];
        a2 += w * aj_w[f];
    }
    s1[i] = a1 + ai_b[0];
    s2[i] = a2 + aj_b[0];
}

// ---------------- kernel 3: fused masked-softmax + P@Wh + ELU ----------------
// WG = 8 waves sharing batch b; TDM double-buffers WhT K-tiles into LDS.
__global__ void __launch_bounds__(256) attn_kernel(const int* __restrict__ adj,
                                                   const half_t* __restrict__ whT,
                                                   const float* __restrict__ s1,
                                                   const float* __restrict__ s2,
                                                   float* __restrict__ out) {
    __shared__ alignas(16) half_t tile0[OUTF_ * 32];
    __shared__ alignas(16) half_t tile1[OUTF_ * 32];

    const int lane = threadIdx.x & 31;
    const int wave = threadIdx.x >> 5;
    const int g    = lane >> 4;
    const int ln   = lane & 15;

    const int rt = blockIdx.x * 8 + wave;   // 8 consecutive row tiles share b
    const int b  = rt >> 7;
    const int n0 = (rt & 127) << 4;
    const int row = n0 + ln;

    const int*    adjRow = adj + ((size_t)b * N_ + row) * N_;
    const float*  s2p    = s2 + b * N_;
    const float   s1v    = s1[b * N_ + row];
    const half_t* whB    = whT + (size_t)b * OUTF_ * N_;

    float m_run = -1e30f, l_run = 0.f;
    v8f acc[16] = {};

    if (wave == 0) stage_issue(tile0, whB, lane);   // prologue: K-block 0 -> tile0

    auto step = [&](const half_t* cur, half_t* nxt, int m0, bool issue_next) {
        // ---- scores for cols m0+g*8+[0..7] and m0+16+g*8+[0..7] (overlaps DMA) ----
        i32x4 aA0 = *(const i32x4*)(adjRow + m0 + g * 8);
        i32x4 aA1 = *(const i32x4*)(adjRow + m0 + g * 8 + 4);
        i32x4 aB0 = *(const i32x4*)(adjRow + m0 + 16 + g * 8);
        i32x4 aB1 = *(const i32x4*)(adjRow + m0 + 16 + g * 8 + 4);
        f32x4 sA0 = *(const f32x4*)(s2p + m0 + g * 8);
        f32x4 sA1 = *(const f32x4*)(s2p + m0 + g * 8 + 4);
        f32x4 sB0 = *(const f32x4*)(s2p + m0 + 16 + g * 8);
        f32x4 sB1 = *(const f32x4*)(s2p + m0 + 16 + g * 8 + 4);
        if (m0 + 32 < N_) __builtin_prefetch(adjRow + m0 + 32, 0, 1);

        float e[16];
        float lm = -1e30f;
#pragma unroll
        for (int j = 0; j < 16; ++j) {
            int   av = (j < 4) ? aA0[j] : (j < 8) ? aA1[j - 4] : (j < 12) ? aB0[j - 8] : aB1[j - 12];
            float sv = (j < 4) ? sA0[j] : (j < 8) ? sA1[j - 4] : (j < 12) ? sB0[j - 8] : sB1[j - 12];
            float ev = s1v + sv;
            ev = (ev >= 0.f) ? ev : ALPHA_ * ev;        // LeakyReLU
            ev = (av > 0) ? ev : NEGINF_;               // adjacency mask
            e[j] = ev;
            lm = fmaxf(lm, ev);
        }

        // ---- publish current tile; kick off next DMA ----
        if (wave == 0) stage_wait();
        __syncthreads();
        if (issue_next && wave == 0) stage_issue(nxt, whB + (m0 + 32), lane);

        // ---- online softmax ----
        lm = fmaxf(lm, __shfl_xor(lm, 16, 32));         // lanes l, l^16 share a row
        float m_new = fmaxf(m_run, lm);
        unsigned long long need = __ballot(m_new > m_run);
        float corr = __expf(m_run - m_new);

        v16h a;
        float psum = 0.f;
#pragma unroll
        for (int j = 0; j < 16; ++j) {
            float p = __expf(e[j] - m_new);
            psum += p;
            a[j] = (half_t)p;                           // element j == A-layout col j
        }
        psum += __shfl_xor(psum, 16, 32);
        l_run = l_run * corr + psum;
        m_run = m_new;

        if (need) {                                     // rescale only when max moved
            float corrR[8];
#pragma unroll
            for (int r = 0; r < 8; ++r) corrR[r] = __shfl(corr, g * 8 + r, 32);
#pragma unroll
            for (int t = 0; t < 16; ++t)
#pragma unroll
                for (int r = 0; r < 8; ++r) acc[t][r] *= corrR[r];
        }

        // ---- 16 feature tiles from LDS: B[k][f], table-order K per lane ----
#pragma unroll
        for (int t = 0; t < 16; ++t) {
            const half_t* bp = cur + (t * 16 + ln) * 32 + g * 8;
            AB32 Bm;
            Bm.u[0] = *(const u32x4*)(bp);
            Bm.u[1] = *(const u32x4*)(bp + 16);
            acc[t] = __builtin_amdgcn_wmma_f32_16x16x32_f16(false, a, false, Bm.h,
                                                            (short)0, acc[t], false, false);
        }
    };

    for (int m0 = 0; m0 < N_; m0 += 64) {
        step(tile0, tile1, m0, true);                   // m0+32 always < N_
        step(tile1, tile0, m0 + 32, (m0 + 64) < N_);
    }

    // ---- epilogue: / softmax denom, ELU, store fp32 ----
    float inv[8];
#pragma unroll
    for (int r = 0; r < 8; ++r) inv[r] = 1.0f / __shfl(l_run, g * 8 + r, 32);
#pragma unroll
    for (int t = 0; t < 16; ++t) {
#pragma unroll
        for (int r = 0; r < 8; ++r) {
            float x = acc[t][r] * inv[r];
            x = (x > 0.f) ? x : (__expf(x) - 1.f);      // ELU, alpha=1
            out[((size_t)(b * N_ + n0 + g * 8 + r)) * OUTF_ + t * 16 + ln] = x;
        }
    }
}

extern "C" void kernel_launch(void* const* d_in, const int* in_sizes, int n_in,
                              void* d_out, int out_size, void* d_ws, size_t ws_size,
                              hipStream_t stream) {
    const float* h    = (const float*)d_in[0];
    const int*   adj  = (const int*)d_in[1];
    const float* W_w  = (const float*)d_in[2];
    const float* W_b  = (const float*)d_in[3];
    const float* ai_w = (const float*)d_in[4];
    const float* ai_b = (const float*)d_in[5];
    const float* aj_w = (const float*)d_in[6];
    const float* aj_b = (const float*)d_in[7];
    float* out = (float*)d_out;

    char* ws = (char*)d_ws;
    const size_t h16_bytes = (size_t)B_ * N_ * INF_ * sizeof(half_t);   // 8 MB
    const size_t w16_bytes = (size_t)OUTF_ * INF_ * sizeof(half_t);     // 128 KB
    const size_t whT_bytes = (size_t)B_ * OUTF_ * N_ * sizeof(half_t);  // 8 MB
    const size_t s_bytes   = (size_t)B_ * N_ * sizeof(float);           // 64 KB

    half_t* h16 = (half_t*)ws;
    half_t* W16 = (half_t*)(ws + h16_bytes);
    half_t* whT = (half_t*)(ws + h16_bytes + w16_bytes);
    float*  s1  = (float*)(ws + h16_bytes + w16_bytes + whT_bytes);
    float*  s2  = (float*)(ws + h16_bytes + w16_bytes + whT_bytes + s_bytes);

    cvt_kernel<<<(B_ * N_ * INF_ + 255) / 256, 256, 0, stream>>>(h, W_w, h16, W16);
    wh_kernel<<<2048, 256, 0, stream>>>(h16, W16, W_b, whT);
    score_kernel<<<(B_ * N_ + 255) / 256, 256, 0, stream>>>(whT, ai_w, ai_b, aj_w, aj_b, s1, s2);
    attn_kernel<<<128, 256, 0, stream>>>(adj, whT, s1, s2, out);
}
